// Network_65051574665498
// MI455X (gfx1250) — matrix operand order repaired and checked by reading, not compiled, not run on hardware
//
#include <hip/hip_runtime.h>
#include <cmath>

#define NPTS 2097152
#define AST 96          // activation row stride in halves (mult of 8 -> 16B aligned rows)
#define NW 2            // waves per block
#define MROWS 32        // rows (points) per wave per iteration: two 16-row subtiles
#define TPW 8           // iterations per wave -> 32*8 = 256 points per wave
#define THREADS (NW * 32)

typedef __attribute__((ext_vector_type(16))) _Float16 v16h;
typedef __attribute__((ext_vector_type(8)))  _Float16 v8h;
typedef __attribute__((ext_vector_type(8)))  float    v8f;

// packed weight offsets in halves: [kc][nt][lane 0..31][half 0..15]
#define OFF_BW0 0
#define OFF_BW1 4096
#define OFF_BW2 8192
#define OFF_HW0 12288
#define OFF_HW1 18432
#define OFF_HW2 22528
#define W_TOTAL 23552   // 47104 bytes

#define WMMA_F16(A, B, C) \
  __builtin_amdgcn_wmma_f32_16x16x32_f16(false, (A), false, (B), (short)0, (C), false, false)

// A-fragment (16x32 f16) from row-major LDS tile: lane l holds row M=l&15;
// lanes 0-15: K in {0..7, 16..23}; lanes 16-31: K in {8..15, 24..31} (ISA 7.12.2)
__device__ __forceinline__ v16h load_a(const _Float16* act, int l, int kc) {
  int m = l & 15, s = l >> 4;
  const _Float16* p = act + m * AST + kc * 32 + s * 8;
  v8h lo = *(const v8h*)p;          // ds_load_b128
  v8h hi = *(const v8h*)(p + 16);   // ds_load_b128
  return __builtin_shufflevector(lo, hi, 0,1,2,3,4,5,6,7,8,9,10,11,12,13,14,15);
}

// Pack global f32 weights [Kreal][Nreal] into LDS B-fragment layout:
// lane l holds column N = nt*16 + (l&15), halves h=0..15 -> K = kc*32 + 8*(l>>4) + (h&7) + 16*(h>>3)
__device__ __forceinline__ void pack_w(_Float16* sw, int dst, const float* W,
                                       int Kreal, int Nreal, int nkc, int nnt, int tid) {
  int total = nkc * nnt * 512;
  for (int idx = tid; idx < total; idx += THREADS) {
    int h  = idx & 15;
    int l  = (idx >> 4) & 31;
    int nt = (idx >> 9) % nnt;
    int kc = idx / (512 * nnt);
    int k = kc * 32 + (l >> 4) * 8 + (h & 7) + (h >> 3) * 16;
    int n = nt * 16 + (l & 15);
    float v = (k < Kreal && n < Nreal) ? W[k * Nreal + n] : 0.0f;
    sw[dst + idx] = (_Float16)v;
  }
}

// In-place dense layer over 32 rows: all A fragments register-resident before
// any store, so input/output share one LDS buffer. Two independent accumulator
// chains (c0/c1) per B fragment hide WMMA->WMMA latency and halve B LDS traffic.
template <int NKC>
__device__ __forceinline__ void dense_relu(_Float16* buf, const _Float16* wB,
                                           const float* bias, int l) {
  v16h a0[NKC], a1[NKC];
#pragma unroll
  for (int kc = 0; kc < NKC; ++kc) {
    a0[kc] = load_a(buf, l, kc);
    a1[kc] = load_a(buf + 16 * AST, l, kc);
  }
  const int nl = l & 15, s = l >> 4;
#pragma unroll
  for (int nt = 0; nt < 4; ++nt) {
    float b = bias[nt * 16 + nl];
    v8f c0, c1;
#pragma unroll
    for (int j = 0; j < 8; ++j) { c0[j] = b; c1[j] = b; }
#pragma unroll
    for (int kc = 0; kc < NKC; ++kc) {
      v16h bf = *(const v16h*)(wB + ((kc * 4 + nt) * 32 + l) * 16);
      c0 = WMMA_F16(a0[kc], bf, c0);
      c1 = WMMA_F16(a1[kc], bf, c1);
    }
#pragma unroll
    for (int j = 0; j < 8; ++j) {
      int r = (j + 8 * s) * AST + nt * 16 + nl;
      buf[r]            = (_Float16)fmaxf(c0[j], 0.0f);
      buf[r + 16 * AST] = (_Float16)fmaxf(c1[j], 0.0f);
    }
  }
}

// amdgpu_waves_per_eu(4): clamp VGPR allocation to <=256 (occupancy is LDS-bound
// anyway); removes the s_set_vgpr_msb churn from allocating VGPRs above 255.
__global__ __launch_bounds__(THREADS)
__attribute__((amdgpu_waves_per_eu(4))) void nerf_fused(
    const float* __restrict__ pos, const float* __restrict__ dir,
    const float* __restrict__ bw0, const float* __restrict__ bb0,
    const float* __restrict__ bw1, const float* __restrict__ bb1,
    const float* __restrict__ bw2, const float* __restrict__ bb2,
    const float* __restrict__ hw0, const float* __restrict__ hb0,
    const float* __restrict__ hw1, const float* __restrict__ hb1,
    const float* __restrict__ hw2, const float* __restrict__ hb2,
    float* __restrict__ out) {
  __shared__ __align__(32) _Float16 s_w[W_TOTAL];
  __shared__ __align__(32) _Float16 s_act[NW][MROWS * AST];
  __shared__ float s_vec[NW][MROWS * 3];

  const int tid = threadIdx.x;
  // one-time cooperative pack of all weights into WMMA B-fragment layout
  pack_w(s_w, OFF_BW0, bw0, 60, 64, 2, 4, tid);
  pack_w(s_w, OFF_BW1, bw1, 64, 64, 2, 4, tid);
  pack_w(s_w, OFF_BW2, bw2, 64, 64, 2, 4, tid);
  pack_w(s_w, OFF_HW0, hw0, 87, 64, 3, 4, tid);
  pack_w(s_w, OFF_HW1, hw1, 64, 64, 2, 4, tid);
  pack_w(s_w, OFF_HW2, hw2, 64,  3, 2, 1, tid);
  __syncthreads();

  const int w = tid >> 5, l = tid & 31;
  const int nl = l & 15, s = l >> 4;
  _Float16* buf = s_act[w];
  float* sv = s_vec[w];
  const float PI  = 3.14159265358979323846f;
  const float HPI = 1.57079632679489661923f;  // cos(x) = sin(x + pi/2): no sin/cos branch

  // head-input pad columns 87..95 are never overwritten -> zero once
  for (int i = l; i < MROWS * 9; i += 32)
    buf[(i / 9) * AST + 87 + (i % 9)] = (_Float16)0.0f;

  for (int it = 0; it < TPW; ++it) {
    const int base = ((blockIdx.x * NW + w) * TPW + it) * MROWS;

    // -------- position frequency encoding -> buf cols 0..63 (60 real + 4 zero)
    for (int i = l; i < MROWS * 3; i += 32) sv[i] = pos[base * 3 + i];
    for (int i = l; i < MROWS * 64; i += 32) {
      int row = i >> 6, k = i & 63;
      float v = 0.0f;
      if (k < 60) {
        int f = k / 6, r = k % 6;
        float ang = sv[row * 3 + (r % 3)] * (PI * (float)(1 << f)) + (r < 3 ? 0.0f : HPI);
        v = __sinf(ang);
      }
      buf[row * AST + k] = (_Float16)v;
    }

    // -------- base MLP (in place)
    dense_relu<2>(buf, s_w + OFF_BW0, bb0, l);
    dense_relu<2>(buf, s_w + OFF_BW1, bb1, l);

    // -------- B2 (linear): col 0 -> density (trunc_exp fwd = exp),
    //          cols 1..63 -> buf cols 24..86 (head input)
    {
      v16h a0[2], a1[2];
#pragma unroll
      for (int kc = 0; kc < 2; ++kc) {
        a0[kc] = load_a(buf, l, kc);
        a1[kc] = load_a(buf + 16 * AST, l, kc);
      }
#pragma unroll
      for (int nt = 0; nt < 4; ++nt) {
        float b = bb2[nt * 16 + nl];
        v8f c0, c1;
#pragma unroll
        for (int j = 0; j < 8; ++j) { c0[j] = b; c1[j] = b; }
#pragma unroll
        for (int kc = 0; kc < 2; ++kc) {
          v16h bf = *(const v16h*)(s_w + OFF_BW2 + ((kc * 4 + nt) * 32 + l) * 16);
          c0 = WMMA_F16(a0[kc], bf, c0);
          c1 = WMMA_F16(a1[kc], bf, c1);
        }
        int n = nt * 16 + nl;
#pragma unroll
        for (int j = 0; j < 8; ++j) {
          int row = j + 8 * s;
          if (n == 0) {
            out[3 * NPTS + base + row]      = __expf(c0[j]);   // density
            out[3 * NPTS + base + 16 + row] = __expf(c1[j]);
          } else {
            buf[row * AST + (23 + n)]        = (_Float16)c0[j];
            buf[(16 + row) * AST + (23 + n)] = (_Float16)c1[j];
          }
        }
      }
    }

    // -------- direction frequency encoding -> buf cols 0..23
    for (int i = l; i < MROWS * 3; i += 32) sv[i] = dir[base * 3 + i];
    for (int i = l; i < MROWS * 24; i += 32) {
      int row = i / 24, k = i % 24;
      int f = k / 6, r = k % 6;
      float ang = sv[row * 3 + (r % 3)] * (PI * (float)(1 << f)) + (r < 3 ? 0.0f : HPI);
      buf[row * AST + k] = (_Float16)__sinf(ang);
    }

    // -------- head MLP (in place; K=96, 87 real)
    dense_relu<3>(buf, s_w + OFF_HW0, hb0, l);
    dense_relu<2>(buf, s_w + OFF_HW1, hb1, l);

    // -------- H2: 64 -> 3, sigmoid -> rgb
    {
      v16h a0[2], a1[2];
#pragma unroll
      for (int kc = 0; kc < 2; ++kc) {
        a0[kc] = load_a(buf, l, kc);
        a1[kc] = load_a(buf + 16 * AST, l, kc);
      }
      float b = (nl < 3) ? hb2[nl] : 0.0f;
      v8f c0, c1;
#pragma unroll
      for (int j = 0; j < 8; ++j) { c0[j] = b; c1[j] = b; }
#pragma unroll
      for (int kc = 0; kc < 2; ++kc) {
        v16h bf = *(const v16h*)(s_w + OFF_HW2 + (kc * 32 + l) * 16);
        c0 = WMMA_F16(a0[kc], bf, c0);
        c1 = WMMA_F16(a1[kc], bf, c1);
      }
      if (nl < 3) {
#pragma unroll
        for (int j = 0; j < 8; ++j) {
          int row = base + j + 8 * s;
          out[row * 3 + nl]        = 1.0f / (1.0f + __expf(-c0[j]));
          out[(row + 16) * 3 + nl] = 1.0f / (1.0f + __expf(-c1[j]));
        }
      }
    }
  }
}

extern "C" void kernel_launch(void* const* d_in, const int* in_sizes, int n_in,
                              void* d_out, int out_size, void* d_ws, size_t ws_size,
                              hipStream_t stream) {
  (void)in_sizes; (void)n_in; (void)out_size; (void)d_ws; (void)ws_size;
  const float* pos = (const float*)d_in[0];
  const float* dir = (const float*)d_in[1];
  const float* bw0 = (const float*)d_in[2];
  const float* bb0 = (const float*)d_in[3];
  const float* bw1 = (const float*)d_in[4];
  const float* bb1 = (const float*)d_in[5];
  const float* bw2 = (const float*)d_in[6];
  const float* bb2 = (const float*)d_in[7];
  const float* hw0 = (const float*)d_in[8];
  const float* hb0 = (const float*)d_in[9];
  const float* hw1 = (const float*)d_in[10];
  const float* hb1 = (const float*)d_in[11];
  const float* hw2 = (const float*)d_in[12];
  const float* hb2 = (const float*)d_in[13];
  float* out = (float*)d_out;

  const int points_per_block = NW * TPW * MROWS;   // 512
  const int blocks = NPTS / points_per_block;      // 4096
  nerf_fused<<<blocks, THREADS, 0, stream>>>(pos, dir, bw0, bb0, bw1, bb1, bw2, bb2,
                                             hw0, hb0, hw1, hb1, hw2, hb2, out);
}
